// QuantizedCompressedLinearLayer_15281493639937
// MI455X (gfx1250) — matrix operand order repaired and checked by reading, not compiled, not run on hardware
//
#include <hip/hip_runtime.h>
#include <hip/hip_bf16.h>

// ---------------------------------------------------------------------------
// QuantizedCompressedLinearLayer for MI455X (gfx1250, wave32, WMMA)
//
//   out[b,s,o] = sum_i x[b,s,i] * (sum_r A[o,r]*B[i,r]) + bias[o]
//   Factored:  T = x @ B   (M=8192,K=4096,N=512)
//              out = T @ A^T + bias   (M=8192,K=512,N=4096)
// ---------------------------------------------------------------------------

typedef __attribute__((ext_vector_type(16))) _Float16 v16h;
typedef __attribute__((ext_vector_type(8)))  _Float16 v8h;
typedef __attribute__((ext_vector_type(8)))  float    v8f;
typedef __attribute__((ext_vector_type(8)))  int      v8i;

#define OUT_DIM 4096
#define IN_DIM  4096
#define RANK    512
#define MROWS   8192   // B*S = 4*2048

// ---------------------------------------------------------------------------
// Kernel 1: convert x fp32 -> f16 (vectorized, 8 elems / thread)
// ---------------------------------------------------------------------------
__global__ __launch_bounds__(256) void cvt_x_kernel(
    const float* __restrict__ x, _Float16* __restrict__ xh, int n8) {
  int i = blockIdx.x * blockDim.x + threadIdx.x;
  if (i >= n8) return;
  v8f v = *((const v8f*)x + i);
  v8h o;
#pragma unroll
  for (int j = 0; j < 8; ++j) o[j] = (_Float16)v[j];
  *((v8h*)xh + i) = o;
}

// ---------------------------------------------------------------------------
// Kernel 2: dequant q_a (int8-valued int32) -> f16, layout preserved [OUT,RANK]
// ---------------------------------------------------------------------------
__global__ __launch_bounds__(256) void dequant_a_kernel(
    const int* __restrict__ q, const float* __restrict__ scale,
    _Float16* __restrict__ out, int n8) {
  int i = blockIdx.x * blockDim.x + threadIdx.x;
  if (i >= n8) return;
  float s = *scale;
  v8i v = *((const v8i*)q + i);
  v8h o;
#pragma unroll
  for (int j = 0; j < 8; ++j) o[j] = (_Float16)((float)v[j] * s);
  *((v8h*)out + i) = o;
}

// ---------------------------------------------------------------------------
// Kernel 3: dequant + transpose q_b [IN,RANK] -> BT f16 [RANK,IN]
// 32x32 tiles through LDS (padded to kill bank conflicts)
// ---------------------------------------------------------------------------
__global__ __launch_bounds__(256) void dequant_bt_kernel(
    const int* __restrict__ q, const float* __restrict__ scale,
    _Float16* __restrict__ BT) {
  __shared__ _Float16 tile[32][33];
  float s = *scale;
  int r0 = blockIdx.x * 32;  // RANK
  int i0 = blockIdx.y * 32;  // IN
  int tx = threadIdx.x;      // 0..31
  int ty = threadIdx.y;      // 0..7
#pragma unroll
  for (int k = 0; k < 4; ++k) {
    int row = ty + k * 8;  // IN offset within tile
    tile[row][tx] = (_Float16)((float)q[(size_t)(i0 + row) * RANK + (r0 + tx)] * s);
  }
  __syncthreads();
#pragma unroll
  for (int k = 0; k < 4; ++k) {
    int rr = ty + k * 8;   // RANK offset within tile
    BT[(size_t)(r0 + rr) * IN_DIM + (i0 + tx)] = tile[tx][rr];
  }
}

// ---------------------------------------------------------------------------
// WMMA fragment loaders (wave32; contiguous-K order, same for A and B)
// A 16x32 f16: lane r=l&15 holds row M=r; h=l>>4 selects contiguous K run {h*16 .. h*16+15}
// B 32x16 f16 (from [N,K] row-major BT): lane c=l&15 is column N=c,
//   h=l>>4 selects contiguous K run {h*16 .. h*16+15}
// ---------------------------------------------------------------------------
__device__ __forceinline__ v16h load_a_frag(const _Float16* __restrict__ A,
                                            int lda, int row, int kb, int lane) {
  int r = lane & 15, h = (lane >> 4) & 1;
  const _Float16* p = A + (size_t)(row + r) * lda + kb + h * 16;
  return *(const v16h*)p;
}

__device__ __forceinline__ v16h load_b_frag(const _Float16* __restrict__ BT,
                                            int ldb, int col, int kb, int lane) {
  int c = lane & 15, h = (lane >> 4) & 1;
  return *(const v16h*)(BT + (size_t)(col + c) * ldb + kb + h * 16);
}

// ---------------------------------------------------------------------------
// Kernel 4: GEMM1  T[8192,512] = xh[8192,4096] @ B  (B via BT[512,4096])
// 128 threads = 4 waves (2x2), wave tile 64x64, block tile 128x128, K-step 32
// ---------------------------------------------------------------------------
__global__ __launch_bounds__(128) void gemm1_kernel(
    const _Float16* __restrict__ xh, const _Float16* __restrict__ BT,
    _Float16* __restrict__ T) {
  const int lane = threadIdx.x & 31;
  const int wave = threadIdx.x >> 5;           // 0..3
  const int mbase = blockIdx.y * 128 + (wave >> 1) * 64;
  const int nbase = blockIdx.x * 128 + (wave & 1) * 64;

  v8f acc[4][4] = {};

  for (int kb = 0; kb < IN_DIM; kb += 32) {
    v16h af[4], bf[4];
#pragma unroll
    for (int mi = 0; mi < 4; ++mi)
      af[mi] = load_a_frag(xh, IN_DIM, mbase + mi * 16, kb, lane);
#pragma unroll
    for (int ni = 0; ni < 4; ++ni)
      bf[ni] = load_b_frag(BT, IN_DIM, nbase + ni * 16, kb, lane);
#pragma unroll
    for (int mi = 0; mi < 4; ++mi)
#pragma unroll
      for (int ni = 0; ni < 4; ++ni)
        acc[mi][ni] = __builtin_amdgcn_wmma_f32_16x16x32_f16(
            false, af[mi], false, bf[ni], (short)0, acc[mi][ni], false, false);
  }

  // D layout: lane c = column, VGPR j = row h*8+j
  const int c = lane & 15, h = (lane >> 4) & 1;
#pragma unroll
  for (int mi = 0; mi < 4; ++mi)
#pragma unroll
    for (int ni = 0; ni < 4; ++ni) {
      int row0 = mbase + mi * 16 + h * 8;
      int col  = nbase + ni * 16 + c;
#pragma unroll
      for (int j = 0; j < 8; ++j)
        T[(size_t)(row0 + j) * RANK + col] = (_Float16)acc[mi][ni][j];
    }
}

// ---------------------------------------------------------------------------
// Kernel 5: GEMM2  out[8192,4096] = T[8192,512] @ A^T + bias
// A_h[OUT,RANK] row-major is already the [N,K] layout GEMM B-fragments want.
// ---------------------------------------------------------------------------
__global__ __launch_bounds__(128) void gemm2_kernel(
    const _Float16* __restrict__ T, const _Float16* __restrict__ Ah,
    const float* __restrict__ bias, float* __restrict__ out) {
  const int lane = threadIdx.x & 31;
  const int wave = threadIdx.x >> 5;
  const int mbase = blockIdx.y * 128 + (wave >> 1) * 64;
  const int nbase = blockIdx.x * 128 + (wave & 1) * 64;

  v8f acc[4][4] = {};

  for (int kb = 0; kb < RANK; kb += 32) {
    v16h af[4], bf[4];
#pragma unroll
    for (int mi = 0; mi < 4; ++mi)
      af[mi] = load_a_frag(T, RANK, mbase + mi * 16, kb, lane);
#pragma unroll
    for (int ni = 0; ni < 4; ++ni)
      bf[ni] = load_b_frag(Ah, RANK, nbase + ni * 16, kb, lane);
#pragma unroll
    for (int mi = 0; mi < 4; ++mi)
#pragma unroll
      for (int ni = 0; ni < 4; ++ni)
        acc[mi][ni] = __builtin_amdgcn_wmma_f32_16x16x32_f16(
            false, af[mi], false, bf[ni], (short)0, acc[mi][ni], false, false);
  }

  const int c = lane & 15, h = (lane >> 4) & 1;
#pragma unroll
  for (int mi = 0; mi < 4; ++mi)
#pragma unroll
    for (int ni = 0; ni < 4; ++ni) {
      int row0 = mbase + mi * 16 + h * 8;
      int col  = nbase + ni * 16 + c;
      float bv = bias[col];
#pragma unroll
      for (int j = 0; j < 8; ++j)
        out[(size_t)(row0 + j) * OUT_DIM + col] = acc[mi][ni][j] + bv;
    }
}

// ---------------------------------------------------------------------------
// Host-side launcher
// ---------------------------------------------------------------------------
extern "C" void kernel_launch(void* const* d_in, const int* in_sizes, int n_in,
                              void* d_out, int out_size, void* d_ws, size_t ws_size,
                              hipStream_t stream) {
  const float* x       = (const float*)d_in[0];   // [8192, 4096] f32
  const int*   q_a     = (const int*)d_in[1];     // [4096, 512]
  const int*   q_b     = (const int*)d_in[2];     // [4096, 512]
  const float* scale_a = (const float*)d_in[3];
  const float* scale_b = (const float*)d_in[4];
  const float* bias    = (const float*)d_in[5];   // [4096]
  float*       out     = (float*)d_out;           // [8192, 4096] f32

  // Workspace layout (all power-of-two aligned)
  char* ws = (char*)d_ws;
  _Float16* xh = (_Float16*)ws;                          // 8192*4096*2 = 64 MiB
  _Float16* Ah = (_Float16*)(ws + (size_t)67108864);     // 4096*512*2  =  4 MiB
  _Float16* BT = (_Float16*)(ws + (size_t)71303168);     // 512*4096*2  =  4 MiB
  _Float16* T  = (_Float16*)(ws + (size_t)75497472);     // 8192*512*2  =  8 MiB

  // 1) x -> f16
  {
    int n8 = (MROWS * IN_DIM) / 8;  // 4,194,304
    cvt_x_kernel<<<n8 / 256, 256, 0, stream>>>(x, xh, n8);
  }
  // 2) dequant A (layout preserved)
  {
    int n8 = (OUT_DIM * RANK) / 8;  // 262,144
    dequant_a_kernel<<<n8 / 256, 256, 0, stream>>>(q_a, scale_a, Ah, n8);
  }
  // 3) dequant + transpose B
  {
    dim3 grid(RANK / 32, IN_DIM / 32);  // 16 x 128
    dim3 block(32, 8);
    dequant_bt_kernel<<<grid, block, 0, stream>>>(q_b, scale_b, BT);
  }
  // 4) T = xh @ B
  {
    dim3 grid(RANK / 128, MROWS / 128);  // 4 x 64
    gemm1_kernel<<<grid, 128, 0, stream>>>(xh, BT, T);
  }
  // 5) out = T @ A^T + bias
  {
    dim3 grid(OUT_DIM / 128, MROWS / 128);  // 32 x 64
    gemm2_kernel<<<grid, 128, 0, stream>>>(T, Ah, bias, out);
  }
}
